// TransformerLayer_81011673137469
// MI455X (gfx1250) — compile-verified, matrix operand order checked by reading, not run on hardware
//
#include <hip/hip_runtime.h>
#include <hip/hip_bf16.h>

// ---------------------------------------------------------------------------
// TransformerLayer for MI455X (gfx1250): all matmuls via v_wmma_f32_16x16x32_f16,
// LDS-staged f16 tiles with register-prefetch pipeline, flash-attention inner
// kernel, conv1d-as-shifted-GEMMs.
// ---------------------------------------------------------------------------

typedef _Float16 v16h __attribute__((ext_vector_type(16)));
typedef _Float16 h8   __attribute__((ext_vector_type(8)));
typedef float    v8f  __attribute__((ext_vector_type(8)));

#define N_HEAD   16
#define D_HEAD   64
#define D_MODEL  1024
#define D_INNER  4096
#define BATCH    4
#define SLEN     1024
#define QKV_LD   3072           // 3 * N_HEAD * D_HEAD
#define NEG_BIG  (-3.0e30f)

__device__ __forceinline__ v8f wmma16(v16h a, v16h b, v8f c) {
  return __builtin_amdgcn_wmma_f32_16x16x32_f16(
      /*neg_a=*/false, a, /*neg_b=*/false, b,
      /*c_mod=*/(short)0, c, /*reuse_a=*/false, /*reuse_b=*/false);
}

// Build a 16-half WMMA fragment from an LDS row.
// Per CDNA5 ISA 16-bit A/B layout: lane group g=0 holds K {0..7,16..23},
// g=1 holds K {8..15,24..31}; chunks are 8 contiguous halfs -> ds_load_b128.
__device__ __forceinline__ v16h lds_frag(const _Float16* rowp, int g) {
  h8 lo = *(const h8*)(rowp + 8 * g);
  h8 hi = *(const h8*)(rowp + 16 + 8 * g);
  return __builtin_shufflevector(lo, hi, 0, 1, 2, 3, 4, 5, 6, 7,
                                 8, 9, 10, 11, 12, 13, 14, 15);
}

// ---------------------------------------------------------------------------
// Generic WMMA GEMM:  C[M,N] (+)= relu( A[M,K] @ B[K,N] + bias )
//   A row-major f32 (lda), optionally row-shifted (conv taps) with per-batch
//   sequence masking (rows crossing a seqMask boundary contribute zeros;
//   loads are UNCONDITIONAL from a clamped in-bounds row, the zeroing is a
//   value select so no exec-mask divergence is generated).
//   B row-major f32 (ldb).  flags: 1=accumulate into C, 2=add bias, 4=ReLU.
// Block: 256 threads (8 waves). Block tile 128x128; wave tile 32x64 (2x4 WMMA).
// Register-prefetch pipeline overlaps next-tile HBM loads with WMMA.
// ---------------------------------------------------------------------------
__global__ __launch_bounds__(256) void gemm_kernel(
    const float* __restrict__ A, int lda, int rowShift, int seqMask,
    const float* __restrict__ Bw, int ldb,
    float* __restrict__ C, int ldc,
    const float* __restrict__ bias, int K, int flags) {
  __shared__ _Float16 As[128][40];   // 128x32 tile, +8 pad: 80B rows, 16B aligned
  __shared__ _Float16 Bs[128][40];   // 128x32 tile stored [n][k]

  const int tid  = threadIdx.x;
  const int lane = tid & 31;
  const int wave = tid >> 5;
  const int g    = lane >> 4;
  const int ln   = lane & 15;
  const int wm   = wave >> 1;        // 0..3 : 32-row slice
  const int wn   = wave & 1;         // 0..1 : 64-col slice

  const int row0 = blockIdx.y * 128;
  const int col0 = blockIdx.x * 128;

  v8f acc[2][4] = {};

  // ---- A staging: 2 threads per row, 16 consecutive f32 each (4x float4).
  const int arow = tid >> 1;
  const int ac0  = (tid & 1) * 16;
  const int gr   = row0 + arow;
  float validf = 1.f;
  int   srow   = gr + rowShift;
  if (seqMask) {
    int local = (gr & (seqMask - 1)) + rowShift;
    if (local < 0 || local >= seqMask) { validf = 0.f; srow = gr; } // clamp in-bounds
  }
  const float* Ap = A + (size_t)srow * lda + ac0;

  // ---- B staging: 2 threads per column, 16 k-values each (strided by ldb).
  const int bn  = tid & 127;
  const int bk0 = (tid >> 7) * 16;   // 0 or 16
  const float* Bp = Bw + (size_t)bk0 * ldb + col0 + bn;

  float4 aReg[4];
  float  bReg[16];

  // prologue: fetch first tile into registers
#pragma unroll
  for (int i = 0; i < 4; ++i) aReg[i] = *(const float4*)(Ap + 4 * i);
#pragma unroll
  for (int i = 0; i < 16; ++i) bReg[i] = Bp[(size_t)i * ldb];

  for (int k0 = 0; k0 < K; k0 += 32) {
    // registers -> LDS (f32 -> f16), zero-select applied to the VALUE only
#pragma unroll
    for (int i = 0; i < 4; ++i) {
      float4 v = aReg[i];
      As[arow][ac0 + 4 * i + 0] = (_Float16)(v.x * validf);
      As[arow][ac0 + 4 * i + 1] = (_Float16)(v.y * validf);
      As[arow][ac0 + 4 * i + 2] = (_Float16)(v.z * validf);
      As[arow][ac0 + 4 * i + 3] = (_Float16)(v.w * validf);
    }
#pragma unroll
    for (int i = 0; i < 16; ++i) Bs[bn][bk0 + i] = (_Float16)bReg[i];
    __syncthreads();

    // prefetch next tile while WMMAs run on this one
    if (k0 + 32 < K) {
#pragma unroll
      for (int i = 0; i < 4; ++i)
        aReg[i] = *(const float4*)(Ap + (k0 + 32) + 4 * i);
#pragma unroll
      for (int i = 0; i < 16; ++i)
        bReg[i] = Bp[(size_t)(k0 + 32 + i) * ldb];
      if (k0 + 64 < K)  // L2 prefetch two tiles ahead on the weight stream
        __builtin_prefetch(Bp + (size_t)(k0 + 64) * ldb, 0, 1);
    }

    v16h a0 = lds_frag(&As[wm * 32 + ln][0], g);
    v16h a1 = lds_frag(&As[wm * 32 + 16 + ln][0], g);
#pragma unroll
    for (int j = 0; j < 4; ++j) {
      v16h bf = lds_frag(&Bs[wn * 64 + j * 16 + ln][0], g);
      acc[0][j] = wmma16(a0, bf, acc[0][j]);
      acc[1][j] = wmma16(a1, bf, acc[1][j]);
    }
    __syncthreads();
  }

  const int rr0 = row0 + wm * 32;
  const int cc0 = col0 + wn * 64;
#pragma unroll
  for (int i = 0; i < 2; ++i)
#pragma unroll
    for (int j = 0; j < 4; ++j)
#pragma unroll
      for (int r = 0; r < 8; ++r) {
        int rr = rr0 + i * 16 + r + 8 * g;
        int cc = cc0 + j * 16 + ln;
        float v = acc[i][j][r];
        if (flags & 2) v += bias[cc];
        if (flags & 1) v += C[(size_t)rr * ldc + cc];
        if (flags & 4) v = fmaxf(v, 0.f);
        C[(size_t)rr * ldc + cc] = v;
      }
}

// ---------------------------------------------------------------------------
// Fused attention (flash-style). Grid: (S/64, B*H). Block: 128 (4 waves).
// Each wave owns 16 query rows; keys streamed in chunks of 32.
// Reads q/k/v in place from the packed QKV buffer [B*S, 3072].
// Writes context already repacked per the reference's h*B+b scatter quirk.
// Additive mask per reference source: pad -> +1.0, valid -> -inf (NEG_BIG),
// with mask batch = bh % B (tf.tile quirk).
// ---------------------------------------------------------------------------
__global__ __launch_bounds__(128) void attn_kernel(
    const float* __restrict__ qkv, const unsigned char* __restrict__ mask,
    float* __restrict__ ctx) {
  const int bh = blockIdx.y;
  const int b  = bh / N_HEAD;
  const int h  = bh % N_HEAD;
  const int bm = bh % BATCH;          // mask batch (tile quirk)
  const int bo = bh % BATCH;          // output scatter quirk
  const int ho = bh / BATCH;

  const int wave = threadIdx.x >> 5;
  const int lane = threadIdx.x & 31;
  const int g    = lane >> 4;
  const int ln   = lane & 15;
  const int q0   = blockIdx.x * 64 + wave * 16;
  const float scale = 0.125f;         // 1/sqrt(64)

  // Q fragments (A-layout), kdim chunks of 32
  v16h qf[2];
#pragma unroll
  for (int c = 0; c < 2; ++c) {
    const float* qp =
        qkv + (size_t)(b * SLEN + q0 + ln) * QKV_LD + h * D_HEAD + c * 32 + 8 * g;
    v16h f = {};
#pragma unroll
    for (int e = 0; e < 8; ++e) f[e] = (_Float16)qp[e];
#pragma unroll
    for (int e = 0; e < 8; ++e) f[8 + e] = (_Float16)qp[16 + e];
    qf[c] = f;
  }

  v8f oacc[4] = {};
  float mrun[8], lrun[8];
#pragma unroll
  for (int r = 0; r < 8; ++r) { mrun[r] = NEG_BIG; lrun[r] = 0.f; }

  __shared__ _Float16 Pt[4][16][40];  // per-wave P re-layout patch

  for (int k0 = 0; k0 < SLEN; k0 += 32) {
    // scores: two 16x16 tiles over key sub-tiles, K=64 in two chunks
    v8f s0 = {}, s1 = {};
#pragma unroll
    for (int c = 0; c < 2; ++c) {
      const float* kp0 = qkv + (size_t)(b * SLEN + k0 + ln) * QKV_LD +
                         D_MODEL + h * D_HEAD + c * 32 + 8 * g;
      const float* kp1 = qkv + (size_t)(b * SLEN + k0 + 16 + ln) * QKV_LD +
                         D_MODEL + h * D_HEAD + c * 32 + 8 * g;
      v16h kb0 = {}, kb1 = {};
#pragma unroll
      for (int e = 0; e < 8; ++e) { kb0[e] = (_Float16)kp0[e];
                                    kb1[e] = (_Float16)kp1[e]; }
#pragma unroll
      for (int e = 0; e < 8; ++e) { kb0[8 + e] = (_Float16)kp0[16 + e];
                                    kb1[8 + e] = (_Float16)kp1[16 + e]; }
      s0 = wmma16(qf[c], kb0, s0);
      s1 = wmma16(qf[c], kb1, s1);
    }

    const float add0 = mask[bm * SLEN + k0 + ln]      ? NEG_BIG : 1.0f;
    const float add1 = mask[bm * SLEN + k0 + 16 + ln] ? NEG_BIG : 1.0f;

    float p0[8], p1[8], alpha[8];
#pragma unroll
    for (int r = 0; r < 8; ++r) {
      p0[r] = s0[r] * scale + add0;
      p1[r] = s1[r] * scale + add1;
      float mx = fmaxf(p0[r], p1[r]);
#pragma unroll
      for (int off = 1; off < 16; off <<= 1)
        mx = fmaxf(mx, __shfl_xor(mx, off, 16));
      float mnew = fmaxf(mrun[r], mx);
      alpha[r] = __expf(mrun[r] - mnew);
      p0[r] = __expf(p0[r] - mnew);
      p1[r] = __expf(p1[r] - mnew);
      float sum = p0[r] + p1[r];
#pragma unroll
      for (int off = 1; off < 16; off <<= 1)
        sum += __shfl_xor(sum, off, 16);
      lrun[r] = lrun[r] * alpha[r] + sum;
      mrun[r] = mnew;
    }
#pragma unroll
    for (int t = 0; t < 4; ++t)
#pragma unroll
      for (int r = 0; r < 8; ++r) oacc[t][r] *= alpha[r];

    // C-layout P -> LDS -> A-layout fragment (wave-private region, no barrier)
#pragma unroll
    for (int r = 0; r < 8; ++r) {
      Pt[wave][r + 8 * g][ln]      = (_Float16)p0[r];
      Pt[wave][r + 8 * g][16 + ln] = (_Float16)p1[r];
    }
    v16h pf = lds_frag(&Pt[wave][ln][0], g);

    // P @ V : four 16-wide d-tiles
#pragma unroll
    for (int t = 0; t < 4; ++t) {
      v16h vb = {};
#pragma unroll
      for (int e = 0; e < 16; ++e) {
        int kk = ((e & 8) ? 16 : 0) + (e & 7) + 8 * g;
        vb[e] = (_Float16)qkv[(size_t)(b * SLEN + k0 + kk) * QKV_LD +
                              2 * D_MODEL + h * D_HEAD + t * 16 + ln];
      }
      oacc[t] = wmma16(pf, vb, oacc[t]);
    }
  }

#pragma unroll
  for (int t = 0; t < 4; ++t)
#pragma unroll
    for (int r = 0; r < 8; ++r) {
      float denom = fmaxf(lrun[r], 1e-20f);
      int q = q0 + r + 8 * g;
      ctx[(size_t)(bo * SLEN + q) * D_MODEL + ho * D_HEAD + t * 16 + ln] =
          oacc[t][r] / denom;
    }
}

// ---------------------------------------------------------------------------
// out[row] = layernorm(a[row] + b[row]) * gamma + beta, then * mask[row].
// One block (256 threads) per row of 1024; single-pass sum/sumsq reduction.
// ---------------------------------------------------------------------------
__global__ __launch_bounds__(256) void ln_mask_kernel(
    const float* __restrict__ A, const float* __restrict__ Bt,
    const float* __restrict__ gamma, const float* __restrict__ beta,
    const unsigned char* __restrict__ mask, float* __restrict__ out) {
  const int row = blockIdx.x;
  const int t   = threadIdx.x;
  const size_t base = (size_t)row * D_MODEL;

  float x[4], s = 0.f, s2 = 0.f;
#pragma unroll
  for (int i = 0; i < 4; ++i) {
    int c = t + i * 256;
    float v = A[base + c] + Bt[base + c];
    x[i] = v; s += v; s2 += v * v;
  }
#pragma unroll
  for (int off = 16; off; off >>= 1) {
    s  += __shfl_xor(s, off, 32);
    s2 += __shfl_xor(s2, off, 32);
  }
  __shared__ float rs[8], rs2[8];
  if ((t & 31) == 0) { rs[t >> 5] = s; rs2[t >> 5] = s2; }
  __syncthreads();
  s = 0.f; s2 = 0.f;
#pragma unroll
  for (int i = 0; i < 8; ++i) { s += rs[i]; s2 += rs2[i]; }
  const float mean = s * (1.f / D_MODEL);
  const float var  = s2 * (1.f / D_MODEL) - mean * mean;
  const float inv  = rsqrtf(var + 1e-3f);
  const float mv   = mask[row] ? 1.f : 0.f;
#pragma unroll
  for (int i = 0; i < 4; ++i) {
    int c = t + i * 256;
    out[base + c] = ((x[i] - mean) * inv * gamma[c] + beta[c]) * mv;
  }
}

// ---------------------------------------------------------------------------
extern "C" void kernel_launch(void* const* d_in, const int* in_sizes, int n_in,
                              void* d_out, int out_size, void* d_ws, size_t ws_size,
                              hipStream_t stream) {
  (void)in_sizes; (void)n_in; (void)out_size; (void)ws_size;

  const float*         dec    = (const float*)d_in[0];
  const unsigned char* mask   = (const unsigned char*)d_in[1];  // bool [B,S,1]
  const float*         qkv_w  = (const float*)d_in[2];
  const float*         qkv_b  = (const float*)d_in[3];
  const float*         o_w    = (const float*)d_in[4];
  const float*         ln1_g  = (const float*)d_in[5];
  const float*         ln1_b  = (const float*)d_in[6];
  const float*         c1w    = (const float*)d_in[7];
  const float*         c1b    = (const float*)d_in[8];
  const float*         c2w    = (const float*)d_in[9];
  const float*         c2b    = (const float*)d_in[10];
  const float*         ln2_g  = (const float*)d_in[11];
  const float*         ln2_b  = (const float*)d_in[12];
  float*               outp   = (float*)d_out;

  const int M = BATCH * SLEN;  // 4096 rows everywhere

  float* ws     = (float*)d_ws;
  float* qkvb   = ws;                 ws += (size_t)M * QKV_LD;   // [4096,3072]
  float* ctx    = ws;                 ws += (size_t)M * D_MODEL;  // [4096,1024]
  float* attn_o = ws;                 ws += (size_t)M * D_MODEL;  // [4096,1024]
  float* out1   = ws;                 ws += (size_t)M * D_MODEL;  // [4096,1024]
  float* core1  = ws;                 ws += (size_t)M * D_INNER;  // [4096,4096]
  float* core2  = ws;                                             // [4096,1024]

  const dim3 blk(256);

  // 1) QKV projection: [4096,1024] @ [1024,3072] + bias
  gemm_kernel<<<dim3(QKV_LD / 128, M / 128), blk, 0, stream>>>(
      dec, D_MODEL, 0, 0, qkv_w, QKV_LD, qkvb, QKV_LD, qkv_b, D_MODEL, /*flags=*/2);

  // 2) Fused attention -> repacked context
  attn_kernel<<<dim3(SLEN / 64, BATCH * N_HEAD), dim3(128), 0, stream>>>(
      qkvb, mask, ctx);

  // 3) Output projection: [4096,1024] @ [1024,1024]
  gemm_kernel<<<dim3(D_MODEL / 128, M / 128), blk, 0, stream>>>(
      ctx, D_MODEL, 0, 0, o_w, D_MODEL, attn_o, D_MODEL, nullptr, D_MODEL, 0);

  // 4) LN1(residual + attn_out) * mask
  ln_mask_kernel<<<dim3(M), blk, 0, stream>>>(dec, attn_o, ln1_g, ln1_b, mask, out1);

  // 5) Conv1 (K=3, SAME) as 3 shifted GEMMs into [4096,4096]; bias first, ReLU last
  for (int k = 0; k < 3; ++k) {
    int flags = (k == 0) ? 2 : 1;
    if (k == 2) flags |= 4;
    gemm_kernel<<<dim3(D_INNER / 128, M / 128), blk, 0, stream>>>(
        out1, D_MODEL, k - 1, SLEN, c1w + (size_t)k * D_MODEL * D_INNER, D_INNER,
        core1, D_INNER, c1b, D_MODEL, flags);
  }

  // 6) Conv2 (K=3, SAME) as 3 shifted GEMMs back to [4096,1024]
  for (int k = 0; k < 3; ++k) {
    int flags = (k == 0) ? 2 : 1;
    gemm_kernel<<<dim3(D_MODEL / 128, M / 128), blk, 0, stream>>>(
        core1, D_INNER, k - 1, SLEN, c2w + (size_t)k * D_INNER * D_MODEL, D_MODEL,
        core2, D_MODEL, c2b, D_INNER, flags);
  }

  // 7) LN2(out1 + conv_out) * mask -> final output
  ln_mask_kernel<<<dim3(M), blk, 0, stream>>>(out1, core2, ln2_g, ln2_b, mask, outp);
}
